// Siege_21964462752572
// MI455X (gfx1250) — matrix operand order repaired and checked by reading, not compiled, not run on hardware
//
#include <hip/hip_runtime.h>
#include <hip/hip_bf16.h>

#define NN 8000
#define NE 80000
#define NG 32
#define C0 128
#define C1 64
#define C2 32
#define NH 4
#define FF 512
#define NBINS 128
#define DW 224           // C0+C1+C2
#define EPSV 1e-6f
#define INV_SQRT_AVG_DEG 0.25335273f    // 1/sqrt(15.57930850982666)
#define INV_SQRT_AVG_NODES 0.23550181f  // 1/sqrt(18.03065905448718)

typedef __attribute__((ext_vector_type(16))) _Float16 v16h;
typedef __attribute__((ext_vector_type(8)))  _Float16 v8h;
typedef __attribute__((ext_vector_type(8)))  float    v8f;

__device__ __forceinline__ float silu_f(float x) { return x / (1.f + __expf(-x)); }
__device__ __forceinline__ float sigm_f(float x) { return 1.f / (1.f + __expf(-x)); }

// ---------------------------------------------------------------------------
// Weight pre-pack: B[K,N] fp32 row-major -> f16 WMMA B-fragment order.
// Tile (kt,nt): K rows kt*32..+31, cols nt*16..+15; 32 lanes x 16 halves,
// tile stride 512 halves (1KB).
// ---------------------------------------------------------------------------
__global__ void k_packB(const float* __restrict__ B, _Float16* __restrict__ P,
                        int N, int K) {
  int idx = blockIdx.x * 256 + threadIdx.x;
  int total = (K >> 5) * (N >> 4) * 32;
  if (idx >= total) return;
  int lane = idx & 31;
  int tile = idx >> 5;
  int ntiles = N >> 4;
  int kt = tile / ntiles, nt = tile - kt * ntiles;
  int col = nt * 16 + (lane & 15);
  int krow = kt * 32 + (lane >> 4) * 8;
  _Float16* dst = P + (size_t)tile * 512 + lane * 16;
#pragma unroll
  for (int e = 0; e < 16; ++e) dst[e] = (_Float16)B[(size_t)(krow + (e & 7) + (e >> 3) * 16) * N + col];
}

// A-fragment loaders (ap pre-offset to row*K + k0 + half*8)
__device__ __forceinline__ v16h load_af(const float* __restrict__ ap) {
  float4 a0 = *(const float4*)(ap);
  float4 a1 = *(const float4*)(ap + 4);
  float4 a2 = *(const float4*)(ap + 16);
  float4 a3 = *(const float4*)(ap + 20);
  v16h a;
  a[0]  = (_Float16)a0.x; a[1]  = (_Float16)a0.y; a[2]  = (_Float16)a0.z; a[3]  = (_Float16)a0.w;
  a[4]  = (_Float16)a1.x; a[5]  = (_Float16)a1.y; a[6]  = (_Float16)a1.z; a[7]  = (_Float16)a1.w;
  a[8]  = (_Float16)a2.x; a[9]  = (_Float16)a2.y; a[10] = (_Float16)a2.z; a[11] = (_Float16)a2.w;
  a[12] = (_Float16)a3.x; a[13] = (_Float16)a3.y; a[14] = (_Float16)a3.z; a[15] = (_Float16)a3.w;
  return a;
}
__device__ __forceinline__ v16h load_af(const _Float16* __restrict__ ap) {
  v8h lo = *(const v8h*)(ap);       // K kb..kb+7   (16B)
  v8h hi = *(const v8h*)(ap + 16);  // K kb+16..+23 (16B)
  v16h a;
#pragma unroll
  for (int e = 0; e < 8; ++e) { a[e] = lo[e]; a[e + 8] = hi[e]; }
  return a;
}

// ---------------------------------------------------------------------------
// Generic WMMA GEMM: C[M,N] = act(A[M,K] @ B[K,N] (+bias)) (+=C)
// A row-major (fp32 or f16), B pre-packed f16 fragments, C fp32 or f16.
// REQUIRES M%64==0, N%32==0, K%32==0. One wave -> 64x32 block of C.
// ---------------------------------------------------------------------------
template<typename AT, typename CT, int ACT, bool ACCUM, bool BIAS>
__global__ __launch_bounds__(32)
void k_gemm(const AT* __restrict__ A, const _Float16* __restrict__ Bp,
            const float* __restrict__ bias, CT* __restrict__ C,
            int N, int K) {
  const int lane = threadIdx.x;
  const int half = lane >> 4;          // 0/1 (K-half selector)
  const int l    = lane & 15;
  const int n0   = blockIdx.x * 32;    // two 16-col tiles
  const int m0r  = blockIdx.y * 64;    // four 16-row tiles
  const int ntiles = N >> 4;

  v8f acc[4][2] = {};
  for (int k0 = 0; k0 < K; k0 += 32) {
    const _Float16* btile = Bp + ((size_t)(k0 >> 5) * ntiles + (n0 >> 4)) * 512 + lane * 16;
    __builtin_prefetch(btile + (size_t)ntiles * 512, 0, 1);
    v16h bf0 = *(const v16h*)(btile);
    v16h bf1 = *(const v16h*)(btile + 512);
    v16h af[4];
#pragma unroll
    for (int t = 0; t < 4; ++t)
      af[t] = load_af(A + (size_t)(m0r + t * 16 + l) * K + k0 + half * 8);
#pragma unroll
    for (int t = 0; t < 4; ++t) {
      acc[t][0] = __builtin_amdgcn_wmma_f32_16x16x32_f16(
          false, af[t], false, bf0, (short)0, acc[t][0], false, false);
      acc[t][1] = __builtin_amdgcn_wmma_f32_16x16x32_f16(
          false, af[t], false, bf1, (short)0, acc[t][1], false, false);
    }
  }
#pragma unroll
  for (int t = 0; t < 4; ++t) {
#pragma unroll
    for (int u = 0; u < 2; ++u) {
      const int col = n0 + u * 16 + l;
#pragma unroll
      for (int j = 0; j < 8; ++j) {
        int row = m0r + t * 16 + half * 8 + j;
        float v = acc[t][u][j];
        if (BIAS) v += bias[col];
        if (ACT == 1) v = silu_f(v);
        CT* cp = C + (size_t)row * N + col;
        if (ACCUM) v += (float)*cp;
        *cp = (CT)v;
      }
    }
  }
}

// ---------------------------------------------------------------------------
// Edge geometry: lengths + spherical harmonics
// ---------------------------------------------------------------------------
__global__ void k_geom(const float* __restrict__ pos, const int* __restrict__ esrc,
                       const int* __restrict__ edst, float* __restrict__ el,
                       float* __restrict__ sh1, float* __restrict__ sh2, int E) {
  int e = blockIdx.x * 256 + threadIdx.x;
  if (e >= E) return;
  int s = esrc[e], d = edst[e];
  float ex = pos[s * 3 + 0] - pos[d * 3 + 0];
  float ey = pos[s * 3 + 1] - pos[d * 3 + 1];
  float ez = pos[s * 3 + 2] - pos[d * 3 + 2];
  float L = sqrtf(ex * ex + ey * ey + ez * ez + EPSV);
  el[e] = L;
  float x = ex / L, y = ey / L, z = ez / L;
  const float s3 = 1.7320508075688772f, s15 = 3.872983346207417f, s5 = 2.23606797749979f;
  sh1[e * 3 + 0] = s3 * x; sh1[e * 3 + 1] = s3 * y; sh1[e * 3 + 2] = s3 * z;
  sh2[e * 5 + 0] = s15 * x * y;
  sh2[e * 5 + 1] = s15 * y * z;
  sh2[e * 5 + 2] = 0.5f * s5 * (3.f * z * z - 1.f);
  sh2[e * 5 + 3] = s15 * x * z;
  sh2[e * 5 + 4] = 0.5f * s15 * (x * x - y * y);
}

__global__ void k_rbf(const float* __restrict__ el, _Float16* __restrict__ eemb, int total) {
  int idx = blockIdx.x * 256 + threadIdx.x;
  if (idx >= total) return;
  int e = idx >> 7, c = idx & 127;
  float mu  = 5.f * (float)c / 127.f;
  float sig = 5.f / 128.f;
  float t = (el[e] - mu) / sig;
  eemb[idx] = (_Float16)__expf(-0.5f * t * t);
}

__global__ void k_embed(const int* __restrict__ f_in, const float* __restrict__ emb,
                        float* __restrict__ x0, int total) {
  int idx = blockIdx.x * 256 + threadIdx.x;
  if (idx >= total) return;
  int n = idx >> 7, c = idx & 127;
  x0[idx] = emb[f_in[n] * C0 + c] * 7.745966692414834f; // sqrt(60)
}

// Initial degree-weighted scatter: w (E x 224) * sh -> x0/x1/x2
__global__ void k_scatter_deg(const float* __restrict__ w, const float* __restrict__ sh1,
                              const float* __restrict__ sh2, const int* __restrict__ edst,
                              float* __restrict__ x0, float* __restrict__ x1,
                              float* __restrict__ x2, int total) {
  int idx = blockIdx.x * 256 + threadIdx.x;
  if (idx >= total) return;
  int e = idx / DW, j = idx - e * DW;
  int d = edst[e];
  float wv = w[(size_t)e * DW + j] * INV_SQRT_AVG_DEG;
  if (j < C0) {
    atomicAdd(&x0[(size_t)d * C0 + j], wv);
  } else if (j < C0 + C1) {
    int c = j - C0;
    for (int m = 0; m < 3; ++m)
      atomicAdd(&x1[(size_t)d * (3 * C1) + m * C1 + c], wv * sh1[e * 3 + m]);
  } else {
    int c = j - C0 - C1;
    for (int m = 0; m < 5; ++m)
      atomicAdd(&x2[(size_t)d * (5 * C2) + m * C2 + c], wv * sh2[e * 5 + m]);
  }
}

// ---------------------------------------------------------------------------
// Equivariant layer norm (wave per node) -> f16 outputs (GEMM inputs)
// ---------------------------------------------------------------------------
__global__ __launch_bounds__(256)
void k_eqln(const float* __restrict__ x0, const float* __restrict__ x1,
            const float* __restrict__ x2, const float* __restrict__ g0,
            const float* __restrict__ b0, const float* __restrict__ g1,
            const float* __restrict__ g2, _Float16* __restrict__ y0,
            _Float16* __restrict__ y1, _Float16* __restrict__ y2, int N) {
  int wave = threadIdx.x >> 5, lane = threadIdx.x & 31;
  int n = blockIdx.x * 8 + wave;
  if (n >= N) return;
  // scalar LN over C0
  const float* xp = x0 + (size_t)n * C0;
  float v[4], s = 0.f, sq = 0.f;
#pragma unroll
  for (int q = 0; q < 4; ++q) { v[q] = xp[lane + q * 32]; s += v[q]; sq += v[q] * v[q]; }
  for (int o = 16; o; o >>= 1) { s += __shfl_xor(s, o); sq += __shfl_xor(sq, o); }
  float m = s * (1.f / C0);
  float rst = rsqrtf(sq * (1.f / C0) - m * m + EPSV);
  _Float16* yp = y0 + (size_t)n * C0;
#pragma unroll
  for (int q = 0; q < 4; ++q) {
    int c = lane + q * 32;
    yp[c] = (_Float16)((v[q] - m) * rst * g0[c] + b0[c]);
  }
  // l=1 norm over 3*C1
  const float* x1p = x1 + (size_t)n * (3 * C1);
  float w1[6]; float q1 = 0.f;
#pragma unroll
  for (int q = 0; q < 6; ++q) { w1[q] = x1p[lane + q * 32]; q1 += w1[q] * w1[q]; }
  for (int o = 16; o; o >>= 1) q1 += __shfl_xor(q1, o);
  float inv1 = rsqrtf(q1 * (1.f / (3.f * C1)) + EPSV);
  _Float16* y1p = y1 + (size_t)n * (3 * C1);
#pragma unroll
  for (int q = 0; q < 6; ++q) {
    int i = lane + q * 32;
    y1p[i] = (_Float16)(w1[q] * inv1 * g1[i & (C1 - 1)]);
  }
  // l=2 norm over 5*C2
  const float* x2p = x2 + (size_t)n * (5 * C2);
  float w2[5]; float q2 = 0.f;
#pragma unroll
  for (int q = 0; q < 5; ++q) { w2[q] = x2p[lane + q * 32]; q2 += w2[q] * w2[q]; }
  for (int o = 16; o; o >>= 1) q2 += __shfl_xor(q2, o);
  float inv2 = rsqrtf(q2 * (1.f / (5.f * C2)) + EPSV);
  _Float16* y2p = y2 + (size_t)n * (5 * C2);
#pragma unroll
  for (int q = 0; q < 5; ++q) {
    int i = lane + q * 32;
    y2p[i] = (_Float16)(w2[q] * inv2 * g2[i & (C2 - 1)]);
  }
}

// scalar LN over F=512 (wave per node) -> f16 output (GEMM input)
__global__ __launch_bounds__(256)
void k_ln512(const float* __restrict__ x, const float* __restrict__ g,
             const float* __restrict__ b, _Float16* __restrict__ y, int N) {
  int wave = threadIdx.x >> 5, lane = threadIdx.x & 31;
  int n = blockIdx.x * 8 + wave;
  if (n >= N) return;
  const float* xp = x + (size_t)n * FF;
  float v[16], s = 0.f, sq = 0.f;
#pragma unroll
  for (int q = 0; q < 16; ++q) { v[q] = xp[lane + q * 32]; s += v[q]; sq += v[q] * v[q]; }
  for (int o = 16; o; o >>= 1) { s += __shfl_xor(s, o); sq += __shfl_xor(sq, o); }
  float m = s * (1.f / FF);
  float rst = rsqrtf(sq * (1.f / FF) - m * m + EPSV);
  _Float16* yp = y + (size_t)n * FF;
#pragma unroll
  for (int q = 0; q < 16; ++q) {
    int c = lane + q * 32;
    yp[c] = (_Float16)((v[q] - m) * rst * g[c] + b[c]);
  }
}

// ---------------------------------------------------------------------------
// Attention: scores (wave per edge), softmax pieces, aggregation
// ---------------------------------------------------------------------------
__device__ __forceinline__ unsigned enc_f(float f) {
  unsigned u = __float_as_uint(f);
  return (u & 0x80000000u) ? ~u : (u | 0x80000000u);
}
__device__ __forceinline__ float dec_f(unsigned u) {
  return (u >> 31) ? __uint_as_float(u ^ 0x80000000u) : __uint_as_float(~u);
}

__global__ void k_fill_u32(unsigned* __restrict__ p, unsigned v, int n) {
  int i = blockIdx.x * 256 + threadIdx.x;
  if (i < n) p[i] = v;
}

__global__ __launch_bounds__(256)
void k_attn_score(const float* __restrict__ s0, const float* __restrict__ t0,
                  const int* __restrict__ esrc, const int* __restrict__ edst,
                  const float* __restrict__ alpha, float* __restrict__ m0,
                  float* __restrict__ araw, unsigned* __restrict__ amax, int E) {
  int wave = threadIdx.x >> 5, lane = threadIdx.x & 31;
  int e = blockIdx.x * 8 + wave;
  if (e >= E) return;
  int s = esrc[e], d = edst[e];
  float pa[NH];
#pragma unroll
  for (int h = 0; h < NH; ++h) {
    int c = h * 32 + lane;
    float mv = s0[(size_t)s * C0 + c] + t0[(size_t)d * C0 + c];
    m0[(size_t)e * C0 + c] = mv;
    pa[h] = mv * alpha[c];
  }
#pragma unroll
  for (int h = 0; h < NH; ++h)
    for (int o = 16; o; o >>= 1) pa[h] += __shfl_xor(pa[h], o);
  if (lane < NH) {
    float a = pa[lane];
    a = a > 0.f ? a : 0.2f * a;       // leaky_relu 0.2
    araw[e * NH + lane] = a;
    atomicMax(&amax[d * NH + lane], enc_f(a));
  }
}

__global__ void k_expsum(const float* __restrict__ araw, const unsigned* __restrict__ amax,
                         const int* __restrict__ edst, float* __restrict__ aexp,
                         float* __restrict__ den, int total) {
  int idx = blockIdx.x * 256 + threadIdx.x;
  if (idx >= total) return;
  int e = idx >> 2, h = idx & 3;
  int d = edst[e];
  float ex = __expf(araw[idx] - dec_f(amax[d * NH + h]));
  aexp[idx] = ex;
  atomicAdd(&den[d * NH + h], ex);
}

__global__ void k_agg0(const float* __restrict__ m0, const float* __restrict__ rw,
                       const float* __restrict__ aexp, const float* __restrict__ den,
                       const int* __restrict__ edst, float* __restrict__ agg0, int total) {
  int idx = blockIdx.x * 256 + threadIdx.x;
  if (idx >= total) return;
  int e = idx >> 7, c = idx & 127;
  int d = edst[e];
  int h = c >> 5;
  float attn = aexp[e * NH + h] / (den[d * NH + h] + EPSV);
  float v = m0[(size_t)e * C0 + c] * rw[(size_t)e * DW + c] * attn;
  atomicAdd(&agg0[(size_t)d * C0 + c], v);
}

__global__ void k_agg1(const float* __restrict__ s1, const float* __restrict__ t1,
                       const float* __restrict__ rw, const float* __restrict__ aexp,
                       const float* __restrict__ den, const int* __restrict__ esrc,
                       const int* __restrict__ edst, float* __restrict__ agg1, int total) {
  int idx = blockIdx.x * 256 + threadIdx.x;
  if (idx >= total) return;
  int e = idx / (3 * C1), j = idx - e * (3 * C1);
  int c = j & (C1 - 1);
  int s = esrc[e], d = edst[e];
  int h = c >> 4;
  float attn = aexp[e * NH + h] / (den[d * NH + h] + EPSV);
  float mv = s1[(size_t)s * (3 * C1) + j] + t1[(size_t)d * (3 * C1) + j];
  atomicAdd(&agg1[(size_t)d * (3 * C1) + j], mv * rw[(size_t)e * DW + C0 + c] * attn);
}

__global__ void k_agg2(const float* __restrict__ s2, const float* __restrict__ t2,
                       const float* __restrict__ rw, const float* __restrict__ aexp,
                       const float* __restrict__ den, const int* __restrict__ esrc,
                       const int* __restrict__ edst, float* __restrict__ agg2, int total) {
  int idx = blockIdx.x * 256 + threadIdx.x;
  if (idx >= total) return;
  int e = idx / (5 * C2), j = idx - e * (5 * C2);
  int c = j & (C2 - 1);
  int s = esrc[e], d = edst[e];
  int h = c >> 3;
  float attn = aexp[e * NH + h] / (den[d * NH + h] + EPSV);
  float mv = s2[(size_t)s * (5 * C2) + j] + t2[(size_t)d * (5 * C2) + j];
  atomicAdd(&agg2[(size_t)d * (5 * C2) + j], mv * rw[(size_t)e * DW + C0 + C1 + c] * attn);
}

// gated FFN nonlinearity: mid0 = silu(h0[:, :128]); mid1/mid2 *= sigmoid(gates)
__global__ void k_gate(const float* __restrict__ h0, _Float16* __restrict__ mid0,
                       _Float16* __restrict__ mid1, _Float16* __restrict__ mid2, int total) {
  int idx = blockIdx.x * 256 + threadIdx.x;
  if (idx >= total) return;
  int n = idx / DW, j = idx - n * DW;
  float hv = h0[idx];
  if (j < C0) {
    mid0[(size_t)n * C0 + j] = (_Float16)silu_f(hv);
  } else if (j < C0 + C1) {
    int c = j - C0;
    float g = sigm_f(hv);
    for (int m = 0; m < 3; ++m) {
      _Float16* p = &mid1[(size_t)n * (3 * C1) + m * C1 + c];
      *p = (_Float16)((float)*p * g);
    }
  } else {
    int c = j - C0 - C1;
    float g = sigm_f(hv);
    for (int m = 0; m < 5; ++m) {
      _Float16* p = &mid2[(size_t)n * (5 * C2) + m * C2 + c];
      *p = (_Float16)((float)*p * g);
    }
  }
}

// final head: out[batch[n]] += (hF[n] . h2 + hb2) / sqrt(AVG_NODES)
__global__ __launch_bounds__(256)
void k_head(const _Float16* __restrict__ hF, const float* __restrict__ h2w,
            const float* __restrict__ hb2, const int* __restrict__ batch,
            float* __restrict__ out, int N) {
  int wave = threadIdx.x >> 5, lane = threadIdx.x & 31;
  int n = blockIdx.x * 8 + wave;
  if (n >= N) return;
  const _Float16* hp = hF + (size_t)n * FF;
  float s = 0.f;
#pragma unroll
  for (int q = 0; q < 16; ++q) s += (float)hp[lane + q * 32] * h2w[lane + q * 32];
  for (int o = 16; o; o >>= 1) s += __shfl_xor(s, o);
  if (lane == 0) atomicAdd(&out[batch[n]], (s + hb2[0]) * INV_SQRT_AVG_NODES);
}

// ---------------------------------------------------------------------------
// Host orchestration
// ---------------------------------------------------------------------------
static inline int cdiv(int a, int b) { return (a + b - 1) / b; }

// GEMM wrappers (AT -> CT, fused op)
static void g_h2h_silu(const _Float16* A, const _Float16* Bp, _Float16* C, int M, int N, int K, hipStream_t st) {
  dim3 g(N / 32, M / 64);
  k_gemm<_Float16, _Float16, 1, false, false><<<g, 32, 0, st>>>(A, Bp, nullptr, C, N, K);
}
static void g_h2h(const _Float16* A, const _Float16* Bp, _Float16* C, int M, int N, int K, hipStream_t st) {
  dim3 g(N / 32, M / 64);
  k_gemm<_Float16, _Float16, 0, false, false><<<g, 32, 0, st>>>(A, Bp, nullptr, C, N, K);
}
static void g_h2f(const _Float16* A, const _Float16* Bp, float* C, int M, int N, int K, hipStream_t st) {
  dim3 g(N / 32, M / 64);
  k_gemm<_Float16, float, 0, false, false><<<g, 32, 0, st>>>(A, Bp, nullptr, C, N, K);
}
static void g_h2f_acc(const _Float16* A, const _Float16* Bp, float* C, int M, int N, int K, hipStream_t st) {
  dim3 g(N / 32, M / 64);
  k_gemm<_Float16, float, 0, true, false><<<g, 32, 0, st>>>(A, Bp, nullptr, C, N, K);
}
static void g_f2f(const float* A, const _Float16* Bp, float* C, int M, int N, int K, hipStream_t st) {
  dim3 g(N / 32, M / 64);
  k_gemm<float, float, 0, false, false><<<g, 32, 0, st>>>(A, Bp, nullptr, C, N, K);
}
static void g_f2f_acc(const float* A, const _Float16* Bp, float* C, int M, int N, int K, hipStream_t st) {
  dim3 g(N / 32, M / 64);
  k_gemm<float, float, 0, true, false><<<g, 32, 0, st>>>(A, Bp, nullptr, C, N, K);
}
static void g_h2h_silu_bias(const _Float16* A, const _Float16* Bp, const float* bias, _Float16* C,
                            int M, int N, int K, hipStream_t st) {
  dim3 g(N / 32, M / 64);
  k_gemm<_Float16, _Float16, 1, false, true><<<g, 32, 0, st>>>(A, Bp, bias, C, N, K);
}

struct BlkPtrs {
  const float *g0, *b0, *g1, *g2, *ws0, *ws1, *ws2, *wd0, *wd1, *wd2;
  const float *fc1, *fc2, *fc3, *alpha;
  const float *wp0, *wp1, *wp2, *g0b, *b0b, *g1b, *g2b;
  const float *f1_0, *f1_1, *f1_2, *f2_0, *f2_1, *f2_2;
  const float *f1, *f2;
};
struct BlkPk {
  const _Float16 *ws0, *ws1, *ws2, *wd0, *wd1, *wd2, *fc1, *fc2, *fc3;
  const _Float16 *wp0, *wp1, *wp2, *f1_0, *f1_1, *f1_2, *f2_0, *f2_1, *f2_2;
  const _Float16 *f1, *f2;
};

extern "C" void kernel_launch(void* const* d_in, const int* in_sizes, int n_in,
                              void* d_out, int out_size, void* d_ws, size_t ws_size,
                              hipStream_t stream) {
  (void)in_sizes; (void)n_in; (void)out_size; (void)ws_size;
  const int*   f_in = (const int*)d_in[0];
  const float* pos  = (const float*)d_in[1];
  const int*   batch = (const int*)d_in[2];
  const int*   esrc = (const int*)d_in[3];
  const int*   edst = (const int*)d_in[4];
  int pi = 5;
  auto nxt = [&]() { return (const float*)d_in[pi++]; };
  const float* atom_embed = nxt();
  const float* dfc1 = nxt(); const float* dfc2 = nxt(); const float* dfc3 = nxt();
  const float* ng = nxt();  const float* nbv = nxt();
  const float* h1w = nxt(); const float* hb1 = nxt();
  const float* h2w = nxt(); const float* hb2 = nxt();
  BlkPtrs bp[6];
  for (int i = 0; i < 6; ++i) {
    BlkPtrs& b = bp[i];
    b.g0 = nxt(); b.b0 = nxt(); b.g1 = nxt(); b.g2 = nxt();
    b.ws0 = nxt(); b.ws1 = nxt(); b.ws2 = nxt();
    b.wd0 = nxt(); b.wd1 = nxt(); b.wd2 = nxt();
    b.fc1 = nxt(); b.fc2 = nxt(); b.fc3 = nxt(); b.alpha = nxt();
    if (i < 5) {
      b.wp0 = nxt(); b.wp1 = nxt(); b.wp2 = nxt();
      b.g0b = nxt(); b.b0b = nxt(); b.g1b = nxt(); b.g2b = nxt();
      b.f1_0 = nxt(); b.f1_1 = nxt(); b.f1_2 = nxt();
      b.f2_0 = nxt(); b.f2_1 = nxt(); b.f2_2 = nxt();
    } else {
      b.wp0 = nxt(); b.g0b = nxt(); b.b0b = nxt(); b.f1 = nxt(); b.f2 = nxt();
    }
  }

  // ---- workspace bump allocator ----
  char* base = (char*)d_ws; size_t off = 0;
  auto alloc = [&](size_t nflt) -> float* {
    float* p = (float*)(base + off);
    off += ((nflt * 4 + 255) / 256) * 256;
    return p;
  };
  auto allocH = [&](size_t nh) -> _Float16* {
    _Float16* p = (_Float16*)(base + off);
    off += ((nh * 2 + 255) / 256) * 256;
    return p;
  };
  auto pack = [&](const float* B, int K, int N) -> const _Float16* {
    _Float16* P = allocH((size_t)K * N);
    int total = (K / 32) * (N / 16) * 32;
    k_packB<<<cdiv(total, 256), 256, 0, stream>>>(B, P, N, K);
    return P;
  };

  float*    el   = alloc((size_t)NE);
  float*    sh1  = alloc((size_t)NE * 3);
  float*    sh2  = alloc((size_t)NE * 5);
  _Float16* eemb = allocH((size_t)NE * NBINS);
  _Float16* rh1  = allocH((size_t)NE * 64);
  _Float16* rh2  = allocH((size_t)NE * 64);
  float*    rw   = alloc((size_t)NE * DW);
  float*    x0   = alloc((size_t)NN * C0);
  float*    x1   = alloc((size_t)NN * 3 * C1);
  float*    x2   = alloc((size_t)NN * 5 * C2);
  _Float16* y0   = allocH((size_t)NN * C0);
  _Float16* y1   = allocH((size_t)NN * 3 * C1);
  _Float16* y2   = allocH((size_t)NN * 5 * C2);
  float*    s0b  = alloc((size_t)NN * C0);
  float*    s1b  = alloc((size_t)NN * 3 * C1);
  float*    s2b  = alloc((size_t)NN * 5 * C2);
  float*    t0b  = alloc((size_t)NN * C0);
  float*    t1b  = alloc((size_t)NN * 3 * C1);
  float*    t2b  = alloc((size_t)NN * 5 * C2);
  float*    m0   = alloc((size_t)NE * C0);
  float*    araw = alloc((size_t)NE * NH);
  float*    aexp = alloc((size_t)NE * NH);
  unsigned* amax = (unsigned*)alloc((size_t)NN * NH);
  float*    den  = alloc((size_t)NN * NH);
  float*    agg0 = alloc((size_t)NN * C0);
  float*    agg1 = alloc((size_t)NN * 3 * C1);
  float*    agg2 = alloc((size_t)NN * 5 * C2);
  float*    h0b  = alloc((size_t)NN * DW);
  _Float16* mid0 = allocH((size_t)NN * C0);
  _Float16* mid1 = allocH((size_t)NN * 3 * C1);
  _Float16* mid2 = allocH((size_t)NN * 5 * C2);
  float*    xF   = alloc((size_t)NN * FF);
  _Float16* zF   = allocH((size_t)NN * FF);
  _Float16* hF   = allocH((size_t)NN * FF);

  // ---- pack all GEMM weights to f16 fragment layout (once per launch) ----
  const _Float16* p_dfc1 = pack(dfc1, NBINS, 64);
  const _Float16* p_dfc2 = pack(dfc2, 64, 64);
  const _Float16* p_dfc3 = pack(dfc3, 64, DW);
  const _Float16* p_h1   = pack(h1w, FF, FF);
  BlkPk pk[6];
  for (int i = 0; i < 6; ++i) {
    const BlkPtrs& b = bp[i];
    BlkPk& k = pk[i];
    k.ws0 = pack(b.ws0, C0, C0); k.wd0 = pack(b.wd0, C0, C0);
    k.ws1 = pack(b.ws1, C1, C1); k.wd1 = pack(b.wd1, C1, C1);
    k.ws2 = pack(b.ws2, C2, C2); k.wd2 = pack(b.wd2, C2, C2);
    k.fc1 = pack(b.fc1, NBINS, 64); k.fc2 = pack(b.fc2, 64, 64); k.fc3 = pack(b.fc3, 64, DW);
    if (i < 5) {
      k.wp0 = pack(b.wp0, C0, C0); k.wp1 = pack(b.wp1, C1, C1); k.wp2 = pack(b.wp2, C2, C2);
      k.f1_0 = pack(b.f1_0, C0, DW); k.f1_1 = pack(b.f1_1, C1, C1); k.f1_2 = pack(b.f1_2, C2, C2);
      k.f2_0 = pack(b.f2_0, C0, C0); k.f2_1 = pack(b.f2_1, C1, C1); k.f2_2 = pack(b.f2_2, C2, C2);
    } else {
      k.wp0 = pack(b.wp0, C0, FF);
      k.f1 = pack(b.f1, FF, FF);
      k.f2 = pack(b.f2, FF, FF);
    }
  }

  float* out = (float*)d_out;
  hipMemsetAsync(out, 0, NG * sizeof(float), stream);

  // ---- edge geometry + RBF features ----
  k_geom<<<cdiv(NE, 256), 256, 0, stream>>>(pos, esrc, edst, el, sh1, sh2, NE);
  k_rbf<<<cdiv(NE * NBINS, 256), 256, 0, stream>>>(el, eemb, NE * NBINS);

  // ---- node embedding + zero l=1/l=2 ----
  k_embed<<<cdiv(NN * C0, 256), 256, 0, stream>>>(f_in, atom_embed, x0, NN * C0);
  hipMemsetAsync(x1, 0, (size_t)NN * 3 * C1 * 4, stream);
  hipMemsetAsync(x2, 0, (size_t)NN * 5 * C2 * 4, stream);

  // ---- radial MLP: eemb -> rw (all f16 until final f32 output) ----
  auto radial = [&](const _Float16* fc1p, const _Float16* fc2p, const _Float16* fc3p) {
    g_h2h_silu(eemb, fc1p, rh1, NE, 64, NBINS, stream);
    g_h2h_silu(rh1, fc2p, rh2, NE, 64, 64, stream);
    g_h2f(rh2, fc3p, rw, NE, DW, 64, stream);
  };
  radial(p_dfc1, p_dfc2, p_dfc3);
  k_scatter_deg<<<cdiv(NE * DW, 256), 256, 0, stream>>>(rw, sh1, sh2, edst, x0, x1, x2, NE * DW);

  const unsigned NEG_INF_ENC = 0x007FFFFFu; // enc_f(-inf)

  for (int i = 0; i < 6; ++i) {
    const BlkPtrs& b = bp[i];
    const BlkPk&   k = pk[i];
    const bool last = (i == 5);

    // equivariant LN (f16 outputs feed GEMMs)
    k_eqln<<<NN / 8, 256, 0, stream>>>(x0, x1, x2, b.g0, b.b0, b.g1, b.g2, y0, y1, y2, NN);

    // source/dest linear maps (f32 outputs feed attention/aggregation)
    g_h2f(y0, k.ws0, s0b, NN, C0, C0, stream);
    g_h2f(y0, k.wd0, t0b, NN, C0, C0, stream);
    g_h2f(y1, k.ws1, s1b, NN * 3, C1, C1, stream);
    g_h2f(y1, k.wd1, t1b, NN * 3, C1, C1, stream);
    g_h2f(y2, k.ws2, s2b, NN * 5, C2, C2, stream);
    g_h2f(y2, k.wd2, t2b, NN * 5, C2, C2, stream);

    // per-block radial weights
    radial(k.fc1, k.fc2, k.fc3);

    // attention softmax over incoming edges
    k_fill_u32<<<cdiv(NN * NH, 256), 256, 0, stream>>>(amax, NEG_INF_ENC, NN * NH);
    hipMemsetAsync(den, 0, (size_t)NN * NH * 4, stream);
    k_attn_score<<<NE / 8, 256, 0, stream>>>(s0b, t0b, esrc, edst, b.alpha, m0, araw, amax, NE);
    k_expsum<<<cdiv(NE * NH, 256), 256, 0, stream>>>(araw, amax, edst, aexp, den, NE * NH);

    // weighted aggregation
    hipMemsetAsync(agg0, 0, (size_t)NN * C0 * 4, stream);
    hipMemsetAsync(agg1, 0, (size_t)NN * 3 * C1 * 4, stream);
    hipMemsetAsync(agg2, 0, (size_t)NN * 5 * C2 * 4, stream);
    k_agg0<<<cdiv(NE * C0, 256), 256, 0, stream>>>(m0, rw, aexp, den, edst, agg0, NE * C0);
    k_agg1<<<cdiv(NE * 3 * C1, 256), 256, 0, stream>>>(s1b, t1b, rw, aexp, den, esrc, edst, agg1, NE * 3 * C1);
    k_agg2<<<cdiv(NE * 5 * C2, 256), 256, 0, stream>>>(s2b, t2b, rw, aexp, den, esrc, edst, agg2, NE * 5 * C2);

    if (!last) {
      // residual projection
      g_f2f_acc(agg0, k.wp0, x0, NN, C0, C0, stream);
      g_f2f_acc(agg1, k.wp1, x1, NN * 3, C1, C1, stream);
      g_f2f_acc(agg2, k.wp2, x2, NN * 5, C2, C2, stream);
      // gated FFN
      k_eqln<<<NN / 8, 256, 0, stream>>>(x0, x1, x2, b.g0b, b.b0b, b.g1b, b.g2b, y0, y1, y2, NN);
      g_h2f(y0, k.f1_0, h0b, NN, DW, C0, stream);
      g_h2h(y1, k.f1_1, mid1, NN * 3, C1, C1, stream);
      g_h2h(y2, k.f1_2, mid2, NN * 5, C2, C2, stream);
      k_gate<<<cdiv(NN * DW, 256), 256, 0, stream>>>(h0b, mid0, mid1, mid2, NN * DW);
      g_h2f_acc(mid0, k.f2_0, x0, NN, C0, C0, stream);
      g_h2f_acc(mid1, k.f2_1, x1, NN * 3, C1, C1, stream);
      g_h2f_acc(mid2, k.f2_2, x2, NN * 5, C2, C2, stream);
    } else {
      // project to F=512, scalar FFN with residual
      g_f2f(agg0, k.wp0, xF, NN, FF, C0, stream);
      k_ln512<<<NN / 8, 256, 0, stream>>>(xF, b.g0b, b.b0b, zF, NN);
      g_h2h_silu(zF, k.f1, hF, NN, FF, FF, stream);
      g_h2f_acc(hF, k.f2, xF, NN, FF, FF, stream);
    }
  }

  // ---- readout head ----
  k_ln512<<<NN / 8, 256, 0, stream>>>(xF, ng, nbv, zF, NN);
  g_h2h_silu_bias(zF, p_h1, hb1, hF, NN, FF, FF, stream);
  k_head<<<NN / 8, 256, 0, stream>>>(hF, h2w, hb2, batch, out, NN);
}
